// PointNetSetAbstraction_26139170963906
// MI455X (gfx1250) — matrix operand order repaired and checked by reading, not compiled, not run on hardware
//
#include <hip/hip_runtime.h>

typedef __attribute__((ext_vector_type(16))) _Float16 v16h;
typedef __attribute__((ext_vector_type(8)))  _Float16 v8h;
typedef __attribute__((ext_vector_type(8)))  float    v8f;

#define B_   16
#define N_   4096
#define C_   64
#define S_   1024
#define K_   32
#define R2_  0.04f
#define EPS_ 1e-5f
#define NG_  (B_ * S_)                 // 16384 groups
#define CNT_ ((float)(B_ * S_ * K_))   // 524288 elements/channel for BN stats

// ---------------------------------------------------------------------------
// helpers
// ---------------------------------------------------------------------------
__device__ __forceinline__ v16h cat8(v8h lo, v8h hi) {
    return __builtin_shufflevector(lo, hi, 0,1,2,3,4,5,6,7,8,9,10,11,12,13,14,15);
}

// A fragment: elements 0-7 = row[k0+8hf .. +7], 8-15 = row[k0+8hf+16 .. +23]
__device__ __forceinline__ v16h load_a(const _Float16* row, int k0, int hf) {
    const v8h lo = *(const v8h*)(row + k0 + 8 * hf);
    const v8h hi = *(const v8h*)(row + k0 + 8 * hf + 16);
    return cat8(lo, hi);
}

// CDNA5 async memory->LDS copy (GLOBAL_LOAD_ASYNC_TO_LDS_B128, ASYNCcnt).
// LDS aperture generic pointers truncate to addr[31:0] = LDS offset.
__device__ __forceinline__ void async_copy_b128(void* lds_ptr, const void* gptr) {
    const unsigned loff = (unsigned)(uintptr_t)lds_ptr;
    asm volatile("global_load_async_to_lds_b128 %0, %1, off"
                 :: "v"(loff), "v"(gptr)
                 : "memory");
}
__device__ __forceinline__ void wait_async() {
    asm volatile("s_wait_asynccnt 0x0" ::: "memory");
}

// BN+ReLU applied in registers to an A fragment; channels for element e are
// cbase + e (e<8) and cbase + 16 + (e-8) (e>=8), cbase = k0 + 8*hf.
__device__ __forceinline__ v16h bn_relu_frag(v16h a, const float* __restrict__ sc,
                                             const float* __restrict__ sh, int cbase) {
    const float4 sA = *(const float4*)(sc + cbase);
    const float4 sB = *(const float4*)(sc + cbase + 4);
    const float4 sC = *(const float4*)(sc + cbase + 16);
    const float4 sD = *(const float4*)(sc + cbase + 20);
    const float4 tA = *(const float4*)(sh + cbase);
    const float4 tB = *(const float4*)(sh + cbase + 4);
    const float4 tC = *(const float4*)(sh + cbase + 16);
    const float4 tD = *(const float4*)(sh + cbase + 20);
    float s[16] = {sA.x,sA.y,sA.z,sA.w, sB.x,sB.y,sB.z,sB.w,
                   sC.x,sC.y,sC.z,sC.w, sD.x,sD.y,sD.z,sD.w};
    float t[16] = {tA.x,tA.y,tA.z,tA.w, tB.x,tB.y,tB.z,tB.w,
                   tC.x,tC.y,tC.z,tC.w, tD.x,tD.y,tD.z,tD.w};
    v16h r;
    #pragma unroll
    for (int e = 0; e < 16; ++e) {
        float v = (float)a[e];
        v = fmaxf(fmaf(v, s[e], t[e]), 0.f);
        r[e] = (_Float16)v;
    }
    return r;
}

// ---------------------------------------------------------------------------
// 0) zero BN-stat accumulators
// ---------------------------------------------------------------------------
__global__ void zero_stats_kernel(float* stats) {
    int t = threadIdx.x;
    for (int i = t; i < 512; i += 256) stats[i] = 0.f;
}

// ---------------------------------------------------------------------------
// 1) Furthest point sampling: one block per batch, dist in LDS, wave32 argmax
// ---------------------------------------------------------------------------
__global__ __launch_bounds__(256) void fps_kernel(const float* __restrict__ p,
                                                  float* __restrict__ qpnt) {
    __shared__ float dist[N_];
    __shared__ float redv[8];
    __shared__ int   redi[8];
    __shared__ int   s_last;
    const int b = blockIdx.x;
    const int t = threadIdx.x;
    const float* pb = p + (size_t)b * N_ * 3;

    for (int i = t; i < N_; i += 256) dist[i] = 1e10f;
    if (t == 0) s_last = 0;
    __syncthreads();

    for (int s = 0; s < S_; ++s) {
        const int last = s_last;
        const float lx = pb[last * 3 + 0];
        const float ly = pb[last * 3 + 1];
        const float lz = pb[last * 3 + 2];
        if (t == 0) {
            float* qp = qpnt + ((size_t)b * S_ + s) * 3;
            qp[0] = lx; qp[1] = ly; qp[2] = lz;
        }
        float bestv = -1.f; int besti = 0;
        for (int i = t; i < N_; i += 256) {
            const float dx = pb[i * 3 + 0] - lx;
            const float dy = pb[i * 3 + 1] - ly;
            const float dz = pb[i * 3 + 2] - lz;
            float dm = fminf(dist[i], dx * dx + dy * dy + dz * dz);
            dist[i] = dm;
            if (dm > bestv) { bestv = dm; besti = i; }
        }
        #pragma unroll
        for (int off = 16; off > 0; off >>= 1) {
            float ov = __shfl_xor(bestv, off, 32);
            int   oi = __shfl_xor(besti, off, 32);
            if (ov > bestv || (ov == bestv && oi < besti)) { bestv = ov; besti = oi; }
        }
        if ((t & 31) == 0) { redv[t >> 5] = bestv; redi[t >> 5] = besti; }
        __syncthreads();
        if (t == 0) {
            float bv = redv[0]; int bi = redi[0];
            #pragma unroll
            for (int w = 1; w < 8; ++w)
                if (redv[w] > bv || (redv[w] == bv && redi[w] < bi)) { bv = redv[w]; bi = redi[w]; }
            s_last = bi;
        }
        __syncthreads();
    }
}

// ---------------------------------------------------------------------------
// 2) Ball query: one wave32 per query; ballot/popc compaction, index order
// ---------------------------------------------------------------------------
__global__ __launch_bounds__(256) void ballq_kernel(const float* __restrict__ p,
                                                    const float* __restrict__ qpnt,
                                                    int* __restrict__ gidx) {
    const int wid  = (blockIdx.x * 256 + threadIdx.x) >> 5;
    const int lane = threadIdx.x & 31;
    const int b    = wid >> 10;
    const float* pb = p + (size_t)b * N_ * 3;
    const float* q  = qpnt + (size_t)wid * 3;
    const float qx = q[0], qy = q[1], qz = q[2];
    int* out = gidx + (size_t)wid * K_;

    int cnt = 0, first = -1;
    for (int base = 0; base < N_ && cnt < K_; base += 32) {
        const int j = base + lane;
        const float dx = pb[j * 3 + 0] - qx;
        const float dy = pb[j * 3 + 1] - qy;
        const float dz = pb[j * 3 + 2] - qz;
        const bool in = (dx * dx + dy * dy + dz * dz) <= R2_;
        const unsigned mask = (unsigned)__ballot(in);
        const int pre  = __popc(mask & ((1u << lane) - 1u));
        const int slot = cnt + pre;
        if (in && slot < K_) out[slot] = j;
        if (first < 0 && mask) first = base + (__ffs(mask) - 1);
        cnt += __popc(mask);
    }
    if (first < 0) first = 0;
    const int filled = cnt < K_ ? cnt : K_;
    if (lane >= filled) out[lane] = first;
}

// ---------------------------------------------------------------------------
// 3) Convert weights to f16 in WMMA B-fragment swizzled order:
//    swz[((ks*NT+n)*32+lane)*16+e] = W[ks*32+8*(lane>>4)+(e&7)+16*(e>>3)][n*16+(lane&15)]
// ---------------------------------------------------------------------------
__device__ __forceinline__ _Float16 swz_elem(const float* __restrict__ W,
                                             int t, int realK, int Cout) {
    const int e    = t & 15;
    const int lane = (t >> 4) & 31;
    const int rest = t >> 9;
    const int NT   = Cout >> 4;
    const int n    = rest % NT;
    const int ks   = rest / NT;
    const int row  = ks * 32 + (lane >> 4) * 8 + (e & 7) + 16 * (e >> 3);
    const int col  = n * 16 + (lane & 15);
    return (_Float16)(row < realK ? W[row * Cout + col] : 0.f);
}

__global__ void cvt_weights_kernel(const float* __restrict__ W0,
                                   const float* __restrict__ W1,
                                   const float* __restrict__ W2,
                                   _Float16* __restrict__ Ws0,
                                   _Float16* __restrict__ Ws1,
                                   _Float16* __restrict__ Ws2) {
    const int t = blockIdx.x * 256 + threadIdx.x;
    if (t < 96 * 64)  Ws0[t] = swz_elem(W0, t, 67, 64);
    if (t < 64 * 64)  Ws1[t] = swz_elem(W1, t, 64, 64);
    if (t < 64 * 128) Ws2[t] = swz_elem(W2, t, 64, 128);
}

// ---------------------------------------------------------------------------
// 4) GEMM0: gather + concat + [32x96]x[96x64] WMMA; raw x0 (f16) + BN stats.
//    Weights staged via async global->LDS; A staged with b128 ds stores.
// ---------------------------------------------------------------------------
__global__ __launch_bounds__(256) void gemm0_kernel(
        const float* __restrict__ f, const float* __restrict__ p,
        const float* __restrict__ qpnt, const int* __restrict__ gidx,
        const _Float16* __restrict__ Wsw, const float* __restrict__ bias,
        _Float16* __restrict__ xout, float* __restrict__ stats) {
    __shared__ alignas(32) _Float16 As[8][32][96];
    __shared__ alignas(32) _Float16 Ws[96 * 64];
    __shared__ float st[128];

    const int t = threadIdx.x, lane = t & 31, w = t >> 5;
    // async-stage swizzled weights into LDS (12KB), tracked with ASYNCcnt
    for (int i = t; i < (96 * 64) / 8; i += 256)
        async_copy_b128(&((v8h*)Ws)[i], &((const v8h*)Wsw)[i]);
    for (int i = t; i < 128; i += 256) st[i] = 0.f;

    const int g = blockIdx.x * 8 + w;
    const int b = g >> 10;
    const float* q = qpnt + (size_t)g * 3;
    const float qx = q[0], qy = q[1], qz = q[2];
    const int idx = gidx[(size_t)g * K_ + lane];
    const float4* f4 = (const float4*)(f + ((size_t)b * N_ + idx) * C_);
    #pragma unroll
    for (int cc = 0; cc < 8; ++cc) {
        const float4 u = f4[cc * 2 + 0];
        const float4 v = f4[cc * 2 + 1];
        v8h h = {(_Float16)u.x, (_Float16)u.y, (_Float16)u.z, (_Float16)u.w,
                 (_Float16)v.x, (_Float16)v.y, (_Float16)v.z, (_Float16)v.w};
        *(v8h*)&As[w][lane][cc * 8] = h;
    }
    const float* pr = p + ((size_t)b * N_ + idx) * 3;
    {
        v8h h = {(_Float16)(pr[0] - qx), (_Float16)(pr[1] - qy), (_Float16)(pr[2] - qz),
                 (_Float16)0.f, (_Float16)0.f, (_Float16)0.f, (_Float16)0.f, (_Float16)0.f};
        *(v8h*)&As[w][lane][64] = h;
        const v8h z8 = {};
        *(v8h*)&As[w][lane][72] = z8;
        *(v8h*)&As[w][lane][80] = z8;
        *(v8h*)&As[w][lane][88] = z8;
    }
    wait_async();
    __syncthreads();

    const int hf = lane >> 4, ln = lane & 15;
    const v8f z = {0.f,0.f,0.f,0.f,0.f,0.f,0.f,0.f};
    v8f acc[2][4];
    #pragma unroll
    for (int m = 0; m < 2; ++m)
        #pragma unroll
        for (int n = 0; n < 4; ++n) acc[m][n] = z;

    #pragma unroll
    for (int ks = 0; ks < 3; ++ks) {
        const int k0 = ks * 32;
        const v16h a0 = load_a(&As[w][ln][0],      k0, hf);
        const v16h a1 = load_a(&As[w][16 + ln][0], k0, hf);
        #pragma unroll
        for (int n = 0; n < 4; ++n) {
            const v8h* wp = (const v8h*)&Ws[((ks * 4 + n) * 32 + lane) * 16];
            const v16h bf = cat8(wp[0], wp[1]);
            acc[0][n] = __builtin_amdgcn_wmma_f32_16x16x32_f16(false, a0, false, bf, (short)0, acc[0][n], false, false);
            acc[1][n] = __builtin_amdgcn_wmma_f32_16x16x32_f16(false, a1, false, bf, (short)0, acc[1][n], false, false);
        }
    }

    #pragma unroll
    for (int n = 0; n < 4; ++n) {
        const int col = n * 16 + ln;
        const float bv = bias[col];
        float s = 0.f, sq = 0.f;
        #pragma unroll
        for (int m = 0; m < 2; ++m)
            #pragma unroll
            for (int r = 0; r < 8; ++r) {
                const float v = acc[m][n][r] + bv;
                const int row = m * 16 + r + 8 * hf;
                xout[((size_t)g * K_ + row) * 64 + col] = (_Float16)v;
                s += v; sq += v * v;
            }
        atomicAdd(&st[col * 2], s);
        atomicAdd(&st[col * 2 + 1], sq);
    }
    __syncthreads();
    for (int i = t; i < 128; i += 256) atomicAdd(&stats[i], st[i]);
}

// ---------------------------------------------------------------------------
// 5) BN finalize
// ---------------------------------------------------------------------------
__global__ void bn_finalize_kernel(const float* __restrict__ stats,
                                   const float* __restrict__ gamma,
                                   const float* __restrict__ beta,
                                   float* __restrict__ scale,
                                   float* __restrict__ shift, int C) {
    const int c = threadIdx.x;
    if (c < C) {
        const float m  = stats[c * 2] / CNT_;
        const float v  = stats[c * 2 + 1] / CNT_ - m * m;
        const float sc = gamma[c] * rsqrtf(v + EPS_);
        scale[c] = sc;
        shift[c] = beta[c] - m * sc;
    }
}

// ---------------------------------------------------------------------------
// 6) GEMM1: A = relu(BN(x0)) in registers on global b128 fragment loads;
//    [32x64]x[64x64] WMMA; raw x1 + BN stats.
// ---------------------------------------------------------------------------
__global__ __launch_bounds__(256) void gemm1_kernel(
        const _Float16* __restrict__ xin,
        const float* __restrict__ scale0, const float* __restrict__ shift0,
        const _Float16* __restrict__ Wsw, const float* __restrict__ bias,
        _Float16* __restrict__ xout, float* __restrict__ stats) {
    __shared__ alignas(32) _Float16 Ws[64 * 64];
    __shared__ float st[128];

    const int t = threadIdx.x, lane = t & 31, w = t >> 5;
    for (int i = t; i < (64 * 64) / 8; i += 256)
        async_copy_b128(&((v8h*)Ws)[i], &((const v8h*)Wsw)[i]);
    for (int i = t; i < 128; i += 256) st[i] = 0.f;
    wait_async();
    __syncthreads();

    const int g = blockIdx.x * 8 + w;
    const int hf = lane >> 4, ln = lane & 15;
    const _Float16* xg = xin + (size_t)g * K_ * 64;

    const v8f z = {0.f,0.f,0.f,0.f,0.f,0.f,0.f,0.f};
    v8f acc[2][4];
    #pragma unroll
    for (int m = 0; m < 2; ++m)
        #pragma unroll
        for (int n = 0; n < 4; ++n) acc[m][n] = z;

    #pragma unroll
    for (int ks = 0; ks < 2; ++ks) {
        const int k0 = ks * 32;
        const int cbase = k0 + 8 * hf;
        v16h a0 = load_a(xg + (size_t)ln * 64,        k0, hf);
        v16h a1 = load_a(xg + (size_t)(16 + ln) * 64, k0, hf);
        a0 = bn_relu_frag(a0, scale0, shift0, cbase);
        a1 = bn_relu_frag(a1, scale0, shift0, cbase);
        #pragma unroll
        for (int n = 0; n < 4; ++n) {
            const v8h* wp = (const v8h*)&Ws[((ks * 4 + n) * 32 + lane) * 16];
            const v16h bf = cat8(wp[0], wp[1]);
            acc[0][n] = __builtin_amdgcn_wmma_f32_16x16x32_f16(false, a0, false, bf, (short)0, acc[0][n], false, false);
            acc[1][n] = __builtin_amdgcn_wmma_f32_16x16x32_f16(false, a1, false, bf, (short)0, acc[1][n], false, false);
        }
    }

    #pragma unroll
    for (int n = 0; n < 4; ++n) {
        const int col = n * 16 + ln;
        const float bv = bias[col];
        float s = 0.f, sq = 0.f;
        #pragma unroll
        for (int m = 0; m < 2; ++m)
            #pragma unroll
            for (int r = 0; r < 8; ++r) {
                const float v = acc[m][n][r] + bv;
                const int row = m * 16 + r + 8 * hf;
                xout[((size_t)g * K_ + row) * 64 + col] = (_Float16)v;
                s += v; sq += v * v;
            }
        atomicAdd(&st[col * 2], s);
        atomicAdd(&st[col * 2 + 1], sq);
    }
    __syncthreads();
    for (int i = t; i < 128; i += 256) atomicAdd(&stats[i], st[i]);
}

// ---------------------------------------------------------------------------
// 7) GEMM2: A = relu(BN(x1)) in registers; [32x64]x[64x128] WMMA; keeps
//    per-(group,channel) max & min over K instead of storing x2.
// ---------------------------------------------------------------------------
__global__ __launch_bounds__(256) void gemm2_kernel(
        const _Float16* __restrict__ xin,
        const float* __restrict__ scale1, const float* __restrict__ shift1,
        const _Float16* __restrict__ Wsw, const float* __restrict__ bias,
        float* __restrict__ maxx, float* __restrict__ minx,
        float* __restrict__ stats) {
    __shared__ alignas(32) _Float16 Ws[64 * 128];
    __shared__ float st[256];

    const int t = threadIdx.x, lane = t & 31, w = t >> 5;
    for (int i = t; i < (64 * 128) / 8; i += 256)
        async_copy_b128(&((v8h*)Ws)[i], &((const v8h*)Wsw)[i]);
    for (int i = t; i < 256; i += 256) st[i] = 0.f;
    wait_async();
    __syncthreads();

    const int g = blockIdx.x * 8 + w;
    const int hf = lane >> 4, ln = lane & 15;
    const _Float16* xg = xin + (size_t)g * K_ * 64;

    const v8f z = {0.f,0.f,0.f,0.f,0.f,0.f,0.f,0.f};
    v8f acc[2][8];
    #pragma unroll
    for (int m = 0; m < 2; ++m)
        #pragma unroll
        for (int n = 0; n < 8; ++n) acc[m][n] = z;

    #pragma unroll
    for (int ks = 0; ks < 2; ++ks) {
        const int k0 = ks * 32;
        const int cbase = k0 + 8 * hf;
        v16h a0 = load_a(xg + (size_t)ln * 64,        k0, hf);
        v16h a1 = load_a(xg + (size_t)(16 + ln) * 64, k0, hf);
        a0 = bn_relu_frag(a0, scale1, shift1, cbase);
        a1 = bn_relu_frag(a1, scale1, shift1, cbase);
        #pragma unroll
        for (int n = 0; n < 8; ++n) {
            const v8h* wp = (const v8h*)&Ws[((ks * 8 + n) * 32 + lane) * 16];
            const v16h bf = cat8(wp[0], wp[1]);
            acc[0][n] = __builtin_amdgcn_wmma_f32_16x16x32_f16(false, a0, false, bf, (short)0, acc[0][n], false, false);
            acc[1][n] = __builtin_amdgcn_wmma_f32_16x16x32_f16(false, a1, false, bf, (short)0, acc[1][n], false, false);
        }
    }

    #pragma unroll
    for (int n = 0; n < 8; ++n) {
        const int col = n * 16 + ln;
        const float bv = bias[col];
        float s = 0.f, sq = 0.f, mx = -3.4e38f, mn = 3.4e38f;
        #pragma unroll
        for (int m = 0; m < 2; ++m)
            #pragma unroll
            for (int r = 0; r < 8; ++r) {
                const float v = acc[m][n][r] + bv;
                s += v; sq += v * v;
                mx = fmaxf(mx, v); mn = fminf(mn, v);
            }
        mx = fmaxf(mx, __shfl_xor(mx, 16, 32));
        mn = fminf(mn, __shfl_xor(mn, 16, 32));
        if (hf == 0) {
            maxx[(size_t)g * 128 + col] = mx;
            minx[(size_t)g * 128 + col] = mn;
        }
        atomicAdd(&st[col * 2], s);
        atomicAdd(&st[col * 2 + 1], sq);
    }
    __syncthreads();
    for (int i = t; i < 256; i += 256) atomicAdd(&stats[i], st[i]);
}

// ---------------------------------------------------------------------------
// 8) Final: f_out = relu((scale>=0 ? max : min) * scale + shift)
// ---------------------------------------------------------------------------
__global__ void out_kernel(const float* __restrict__ maxx, const float* __restrict__ minx,
                           const float* __restrict__ scale, const float* __restrict__ shift,
                           float* __restrict__ fout) {
    const int tid = blockIdx.x * 256 + threadIdx.x;
    if (tid < B_ * S_ * 128) {
        const int ch = tid & 127;
        const float sc = scale[ch];
        const float base = sc >= 0.f ? maxx[tid] : minx[tid];
        fout[tid] = fmaxf(fmaf(base, sc, shift[ch]), 0.f);
    }
}

// ---------------------------------------------------------------------------
extern "C" void kernel_launch(void* const* d_in, const int* in_sizes, int n_in,
                              void* d_out, int out_size, void* d_ws, size_t ws_size,
                              hipStream_t stream) {
    const float* f      = (const float*)d_in[0];
    const float* p      = (const float*)d_in[1];
    const float* W0     = (const float*)d_in[2];
    const float* b0     = (const float*)d_in[3];
    const float* gamma0 = (const float*)d_in[4];
    const float* beta0  = (const float*)d_in[5];
    const float* W1     = (const float*)d_in[6];
    const float* b1     = (const float*)d_in[7];
    const float* gamma1 = (const float*)d_in[8];
    const float* beta1  = (const float*)d_in[9];
    const float* W2     = (const float*)d_in[10];
    const float* b2     = (const float*)d_in[11];
    const float* gamma2 = (const float*)d_in[12];
    const float* beta2  = (const float*)d_in[13];

    float* fout = (float*)d_out;                 // [16,1024,128]
    float* qpnt = fout + (size_t)B_ * S_ * 128;  // [16,1024,3]

    char* ws = (char*)d_ws;
    size_t off = 0;
    auto carve = [&](size_t bytes) -> char* {
        char* r = ws + off;
        off = (off + bytes + 255) & ~(size_t)255;
        return r;
    };
    int*      gidx  = (int*)     carve((size_t)NG_ * K_ * 4);
    _Float16* Ws0   = (_Float16*)carve(96 * 64 * 2);
    _Float16* Ws1   = (_Float16*)carve(64 * 64 * 2);
    _Float16* Ws2   = (_Float16*)carve(64 * 128 * 2);
    float*    stats = (float*)   carve(512 * 4);
    float*    st0 = stats, *st1 = stats + 128, *st2 = stats + 256;
    float*    ssbuf = (float*)   carve(512 * 4);
    float*    sc0 = ssbuf, *sh0 = ssbuf + 64, *sc1 = ssbuf + 128, *sh1 = ssbuf + 192;
    float*    sc2 = ssbuf + 256, *sh2 = ssbuf + 384;
    _Float16* x0    = (_Float16*)carve((size_t)NG_ * K_ * 64 * 2);
    _Float16* x1    = (_Float16*)carve((size_t)NG_ * K_ * 64 * 2);
    float*    maxx  = (float*)   carve((size_t)NG_ * 128 * 4);
    float*    minx  = (float*)   carve((size_t)NG_ * 128 * 4);

    zero_stats_kernel<<<1, 256, 0, stream>>>(stats);
    fps_kernel<<<B_, 256, 0, stream>>>(p, qpnt);
    ballq_kernel<<<NG_ / 8, 256, 0, stream>>>(p, qpnt, gidx);
    cvt_weights_kernel<<<(64 * 128 + 255) / 256, 256, 0, stream>>>(W0, W1, W2, Ws0, Ws1, Ws2);

    gemm0_kernel<<<NG_ / 8, 256, 0, stream>>>(f, p, qpnt, gidx, Ws0, b0, x0, st0);
    bn_finalize_kernel<<<1, 256, 0, stream>>>(st0, gamma0, beta0, sc0, sh0, 64);

    gemm1_kernel<<<NG_ / 8, 256, 0, stream>>>(x0, sc0, sh0, Ws1, b1, x1, st1);
    bn_finalize_kernel<<<1, 256, 0, stream>>>(st1, gamma1, beta1, sc1, sh1, 64);

    gemm2_kernel<<<NG_ / 8, 256, 0, stream>>>(x1, sc1, sh1, Ws2, b2, maxx, minx, st2);
    bn_finalize_kernel<<<1, 256, 0, stream>>>(st2, gamma2, beta2, sc2, sh2, 128);

    out_kernel<<<(B_ * S_ * 128 + 255) / 256, 256, 0, stream>>>(maxx, minx, sc2, sh2, fout);
}
